// BeansAttentionBlock_32547262169460
// MI455X (gfx1250) — compile-verified, matrix operand order checked by reading, not run on hardware
//
#include <hip/hip_runtime.h>
#include <hip/hip_bf16.h>
#include <math.h>
#include <stdint.h>

// ---------------------------------------------------------------------------
// Types for CDNA5 WMMA (wave32): A/B = 16 x bf16, C/D = 8 x f32
// ---------------------------------------------------------------------------
typedef __attribute__((ext_vector_type(16))) __bf16 v16bf;
typedef __attribute__((ext_vector_type(8)))  __bf16 v8bf;
typedef __attribute__((ext_vector_type(8)))  float  v8f;
typedef __attribute__((ext_vector_type(4)))  int    v4i;

typedef __attribute__((address_space(1))) v4i as1_v4i;   // global
typedef __attribute__((address_space(3))) v4i as3_v4i;   // LDS

// gfx1250 async copy to LDS (ASYNCcnt path) if the toolchain declares it.
#if defined(__has_builtin)
#if __has_builtin(__builtin_amdgcn_global_load_async_to_lds_b128) && \
    __has_builtin(__builtin_amdgcn_s_wait_asynccnt)
#define USE_ASYNC_LDS 1
#endif
#endif
#ifndef USE_ASYNC_LDS
#define USE_ASYNC_LDS 0
#endif

__device__ __forceinline__ __bf16 f2bf(float f) {
    union { float f; unsigned u; } in; in.f = f;
    unsigned u = in.u + 0x7FFFu + ((in.u >> 16) & 1u);   // round-to-nearest-even
    union { unsigned short s; __bf16 b; } out; out.s = (unsigned short)(u >> 16);
    return out.b;
}

// 16B global -> LDS copy: async (bypasses VGPRs, ASYNCcnt) or direct fallback.
__device__ __forceinline__ void ld_g2l_b128(const __bf16* g, __bf16* l) {
#if USE_ASYNC_LDS
    __builtin_amdgcn_global_load_async_to_lds_b128(
        (as1_v4i*)(uintptr_t)g,
        (as3_v4i*)(uint32_t)(uintptr_t)l,
        0, 0);
#else
    *(v8bf*)l = *(const v8bf*)g;
#endif
}

__device__ __forceinline__ void g2l_join() {
#if USE_ASYNC_LDS
    __builtin_amdgcn_s_wait_asynccnt(0);
#endif
}

// ---------------------------------------------------------------------------
// Elementwise f32 -> bf16 cast
// ---------------------------------------------------------------------------
__global__ void cast_bf16_kernel(const float* __restrict__ in, __bf16* __restrict__ out, long n) {
    long i = (long)blockIdx.x * 256 + threadIdx.x;
    if (i < n) out[i] = f2bf(in[i]);
}

// f32 W[K,N] -> bf16 WT[N,K] (one-time; output-coalesced)
__global__ void cast_transpose_bf16_kernel(const float* __restrict__ w, __bf16* __restrict__ wt,
                                           int K, int N) {
    long i = (long)blockIdx.x * 256 + threadIdx.x;
    long total = (long)K * N;
    if (i < total) {
        int n = (int)(i / K);
        int k = (int)(i % K);
        wt[i] = f2bf(w[(long)k * N + n]);
    }
}

// ---------------------------------------------------------------------------
// LayerNorm f32 -> bf16, one block per row; pad rows (>= Mreal) write zeros.
// ---------------------------------------------------------------------------
__global__ __launch_bounds__(256)
void ln_bf16_kernel(const float* __restrict__ x, const float* __restrict__ g,
                    const float* __restrict__ b, __bf16* __restrict__ out,
                    int Mreal, int D) {
    int row = blockIdx.x;
    int tid = threadIdx.x;
    if (row >= Mreal) {
        for (int d = tid; d < D; d += 256) out[(long)row * D + d] = f2bf(0.f);
        return;
    }
    __shared__ float s1[256], s2[256];
    const float* xr = x + (long)row * D;
    float a = 0.f, q = 0.f;
    for (int d = tid; d < D; d += 256) { float v = xr[d]; a += v; q += v * v; }
    s1[tid] = a; s2[tid] = q;
    __syncthreads();
    for (int s = 128; s > 0; s >>= 1) {
        if (tid < s) { s1[tid] += s1[tid + s]; s2[tid] += s2[tid + s]; }
        __syncthreads();
    }
    float mean = s1[0] / (float)D;
    float var  = s2[0] / (float)D - mean * mean;
    float inv  = rsqrtf(var + 1e-5f);
    for (int d = tid; d < D; d += 256) {
        float v = (xr[d] - mean) * inv * g[d] + b[d];
        out[(long)row * D + d] = f2bf(v);
    }
}

__global__ void zero_bf16_kernel(__bf16* __restrict__ p, long n) {
    long i = (long)blockIdx.x * 256 + threadIdx.x;
    if (i < n) p[i] = f2bf(0.f);
}

// ---------------------------------------------------------------------------
// WMMA bf16 GEMM: out[M,N] = A[M,K] @ W[K,N] (+bias)(+resid)(gelu)
// W is supplied PRE-TRANSPOSED as WT[N,K] so both A and B tiles stage to LDS
// with identical coalesced b128 copies (async-to-LDS when available) and all
// fragments are pairs of ds_read_b128 matching the ISA 16-bit A/B layouts.
// Block: 256 thr = 8 waves as 4(M)x2(N); block tile 128x128; K-step 32.
// Each wave: 32x64 strip = 2 A frags x 4 B frags = 8 v_wmma per K-step.
// ---------------------------------------------------------------------------
#define GEMM_BM  128
#define GEMM_BN  128
#define GEMM_BK  32
#define GEMM_LDS 48   // padded LDS row stride in bf16 elems (96B, 16B-aligned)

__global__ __launch_bounds__(256)
void gemm_wmma_kernel(const __bf16* __restrict__ A, const __bf16* __restrict__ WT,
                      const float* __restrict__ bias, const float* __restrict__ resid,
                      float* __restrict__ outF, __bf16* __restrict__ outB,
                      int Mstore, int K, int N, int dogelu) {
    __shared__ __bf16 As[GEMM_BM * GEMM_LDS];
    __shared__ __bf16 Ws[GEMM_BN * GEMM_LDS];

    const int tid  = threadIdx.x;
    const int wave = tid >> 5;
    const int lane = tid & 31;
    const int wm   = wave >> 1;          // 0..3 -> M sub-strip of 32 rows
    const int wn   = wave & 1;           // 0..1 -> N sub-strip of 64 cols
    const int row0 = blockIdx.y * GEMM_BM;
    const int n0   = blockIdx.x * GEMM_BN;

    v8f acc[2][4];
#pragma unroll
    for (int m = 0; m < 2; ++m)
#pragma unroll
        for (int t = 0; t < 4; ++t)
#pragma unroll
            for (int i = 0; i < 8; ++i) acc[m][t][i] = 0.f;

    const int r    = lane & 15;
    const int koff = (lane >> 4) ? 8 : 0;   // lanes 16-31 hold K 8..15 / 24..31

    for (int k0 = 0; k0 < K; k0 += GEMM_BK) {
        // ---- stage A and WT tiles: each 128 rows x 32 bf16, b128 chunks ----
#pragma unroll
        for (int cc = 0; cc < 2; ++cc) {
            int c  = tid * 2 + cc;          // 0..511
            int rr = c >> 2;                // 0..127
            int kc = (c & 3) * 8;           // 0,8,16,24
            ld_g2l_b128(A  + (long)(row0 + rr) * K + k0 + kc, As + rr * GEMM_LDS + kc);
            ld_g2l_b128(WT + (long)(n0   + rr) * K + k0 + kc, Ws + rr * GEMM_LDS + kc);
        }
        g2l_join();
        __syncthreads();

        // ---- A fragments: this wave's two 16-row sub-tiles ----
        v16bf afr[2];
#pragma unroll
        for (int m = 0; m < 2; ++m) {
            v8bf lo = *(const v8bf*)(As + (wm * 32 + m * 16 + r) * GEMM_LDS + koff);
            v8bf hi = *(const v8bf*)(As + (wm * 32 + m * 16 + r) * GEMM_LDS + koff + 16);
#pragma unroll
            for (int i = 0; i < 8; ++i) { afr[m][i] = lo[i]; afr[m][i + 8] = hi[i]; }
        }
        // ---- 4 N-subtiles, each B frag reused by both A frags ----
#pragma unroll
        for (int t = 0; t < 4; ++t) {
            v16bf bfr;
            v8bf lo = *(const v8bf*)(Ws + (wn * 64 + t * 16 + r) * GEMM_LDS + koff);
            v8bf hi = *(const v8bf*)(Ws + (wn * 64 + t * 16 + r) * GEMM_LDS + koff + 16);
#pragma unroll
            for (int i = 0; i < 8; ++i) { bfr[i] = lo[i]; bfr[i + 8] = hi[i]; }
            acc[0][t] = __builtin_amdgcn_wmma_f32_16x16x32_bf16(
                false, afr[0], false, bfr, (short)0, acc[0][t], false, false);
            acc[1][t] = __builtin_amdgcn_wmma_f32_16x16x32_bf16(
                false, afr[1], false, bfr, (short)0, acc[1][t], false, false);
        }
        __syncthreads();
    }

    // ---- epilogue: lane l holds N = (l&15), M = j + 8*(l>=16), j=0..7 ----
    const int msub = (lane >> 4) ? 8 : 0;
#pragma unroll
    for (int m = 0; m < 2; ++m) {
#pragma unroll
        for (int t = 0; t < 4; ++t) {
            int col  = n0 + wn * 64 + t * 16 + r;
            float bv = bias ? bias[col] : 0.f;
#pragma unroll
            for (int j = 0; j < 8; ++j) {
                int row = row0 + wm * 32 + m * 16 + msub + j;
                if (row < Mstore) {
                    float v = acc[m][t][j] + bv;
                    if (resid)  v += resid[(long)row * N + col];
                    if (dogelu) v = 0.5f * v * (1.f + erff(v * 0.70710678118654752f));
                    if (outF) outF[(long)row * N + col] = v;
                    if (outB) outB[(long)row * N + col] = f2bf(v);
                }
            }
        }
    }
}

// ---------------------------------------------------------------------------
// CLS attention: one block (128 thr) per (b,h). Full softmax over S keys.
// qkv layout: row = b*S + t, cols [0,768)=Q, [768,1536)=K, [1536,2304)=V,
// within each: h*64 + hd.
// ---------------------------------------------------------------------------
__global__ __launch_bounds__(128)
void attn_cls_kernel(const float* __restrict__ qkv, __bf16* __restrict__ attn,
                     int S, int H) {
    const int b = blockIdx.x / H, h = blockIdx.x % H;
    const int tid = threadIdx.x;
    __shared__ float sc[1025];
    __shared__ float red[128];

    const float* q = qkv + (long)(b * S) * 2304 + h * 64;

    float lmax = -1e30f;
    for (int t = tid; t < S; t += 128) {
        const float* k = qkv + (long)(b * S + t) * 2304 + 768 + h * 64;
        float s = 0.f;
#pragma unroll 8
        for (int d = 0; d < 64; ++d) s += q[d] * k[d];
        s *= 0.125f;
        sc[t] = s;
        lmax = fmaxf(lmax, s);
    }
    red[tid] = lmax; __syncthreads();
    for (int s = 64; s > 0; s >>= 1) {
        if (tid < s) red[tid] = fmaxf(red[tid], red[tid + s]);
        __syncthreads();
    }
    float mx = red[0]; __syncthreads();

    float lsum = 0.f;
    for (int t = tid; t < S; t += 128) {
        float e = expf(sc[t] - mx);
        sc[t] = e; lsum += e;
    }
    red[tid] = lsum; __syncthreads();
    for (int s = 64; s > 0; s >>= 1) {
        if (tid < s) red[tid] += red[tid + s];
        __syncthreads();
    }
    float inv = 1.f / red[0];
    __syncthreads();

    if (tid < 64) {
        float acc = 0.f;
        for (int t = 0; t < S; ++t)
            acc += sc[t] * qkv[(long)(b * S + t) * 2304 + 1536 + h * 64 + tid];
        attn[(long)(b * S) * 768 + h * 64 + tid] = f2bf(acc * inv);
    }
}

// ---------------------------------------------------------------------------
// Routed attention: one wave32 per (b,h,p); 16 gathered keys of dim 64.
// ---------------------------------------------------------------------------
__global__ __launch_bounds__(256)
void attn_routed_kernel(const float* __restrict__ qkv, const int* __restrict__ routes,
                        __bf16* __restrict__ attn, int Bn, int S, int H, int P) {
    const int wid  = (blockIdx.x * 256 + threadIdx.x) >> 5;
    const int lane = threadIdx.x & 31;
    const int total = Bn * H * P;
    if (wid >= total) return;

    const int b   = wid / (H * P);
    const int rem = wid % (H * P);
    const int h   = rem / P;
    const int p   = rem % P;

    const long qrow = (long)(b * S + 1 + p);
    const float* q  = qkv + qrow * 2304 + h * 64;

    const int j    = lane >> 1;
    const int half = lane & 1;
    const int ktok = routes[p * 16 + j] + 1;
    const float* kp = qkv + (long)(b * S + ktok) * 2304 + 768 + h * 64 + half * 32;
    const float* qp = q + half * 32;

    float partial = 0.f;
#pragma unroll 8
    for (int i = 0; i < 32; ++i) partial += qp[i] * kp[i];
    float s = (partial + __shfl_xor(partial, 1, 32)) * 0.125f;

    float m = s;
    for (int o = 16; o > 0; o >>= 1) m = fmaxf(m, __shfl_xor(m, o, 32));
    float e = expf(s - m);
    float c = half ? 0.f : e;                       // count each key once
    float sum = c;
    for (int o = 16; o > 0; o >>= 1) sum += __shfl_xor(sum, o, 32);
    float prob = e / sum;

    float acc0 = 0.f, acc1 = 0.f;
#pragma unroll
    for (int jj = 0; jj < 16; ++jj) {
        float pj = __shfl(prob, jj * 2, 32);
        int kt = routes[p * 16 + jj] + 1;           // uniform across wave
        const float* vp = qkv + (long)(b * S + kt) * 2304 + 1536 + h * 64;
        acc0 += pj * vp[lane];
        acc1 += pj * vp[lane + 32];
    }
    __bf16* o = attn + qrow * 768 + h * 64;
    o[lane]      = f2bf(acc0);
    o[lane + 32] = f2bf(acc1);
}

// ---------------------------------------------------------------------------
// Host-side orchestration
// ---------------------------------------------------------------------------
extern "C" void kernel_launch(void* const* d_in, const int* in_sizes, int n_in,
                              void* d_out, int out_size, void* d_ws, size_t ws_size,
                              hipStream_t stream) {
    (void)in_sizes; (void)n_in; (void)out_size; (void)ws_size;

    const int Bn = 4, P = 1024, D = 768, H = 12;
    const int S = P + 1;                 // 1025
    const int M = Bn * S;                // 4100 tokens
    const int Mpad = 4224;               // 33 * 128

    const float* x      = (const float*)d_in[0];
    const int*   routes = (const int*)  d_in[1];
    const float* qkv_w  = (const float*)d_in[2];
    const float* qkv_b  = (const float*)d_in[3];
    const float* proj_w = (const float*)d_in[4];
    const float* proj_b = (const float*)d_in[5];
    const float* ln1_g  = (const float*)d_in[6];
    const float* ln1_b  = (const float*)d_in[7];
    const float* ln2_g  = (const float*)d_in[8];
    const float* ln2_b  = (const float*)d_in[9];
    const float* mlp_w1 = (const float*)d_in[10];
    const float* mlp_b1 = (const float*)d_in[11];
    const float* mlp_w2 = (const float*)d_in[12];
    const float* mlp_b2 = (const float*)d_in[13];
    float* out = (float*)d_out;

    // workspace carve-up (256B aligned)
    char* base = (char*)d_ws;
    size_t off = 0;
    auto alloc = [&](size_t bytes) -> char* {
        char* p = base + off;
        off += (bytes + 255) & ~(size_t)255;
        return p;
    };
    __bf16* qkv_wT  = (__bf16*)alloc((size_t)D * 3 * D * 2);   // [3D, D]
    __bf16* proj_wT = (__bf16*)alloc((size_t)D * D * 2);       // [D, D]
    __bf16* mlp_w1T = (__bf16*)alloc((size_t)D * 4 * D * 2);   // [4D, D]
    __bf16* mlp_w2T = (__bf16*)alloc((size_t)4 * D * D * 2);   // [D, 4D]
    __bf16* xn      = (__bf16*)alloc((size_t)Mpad * D * 2);
    float*  qkvbuf  = (float*) alloc((size_t)Mpad * 3 * D * 4);
    __bf16* attnb   = (__bf16*)alloc((size_t)Mpad * D * 2);
    float*  x1      = (float*) alloc((size_t)Mpad * D * 4);
    __bf16* h2in    = (__bf16*)alloc((size_t)Mpad * D * 2);
    __bf16* h1      = (__bf16*)alloc((size_t)Mpad * 4 * D * 2);

    auto castT = [&](const float* w, __bf16* wt, int K, int N) {
        long n = (long)K * N;
        cast_transpose_bf16_kernel<<<(int)((n + 255) / 256), 256, 0, stream>>>(w, wt, K, N);
    };

    // 1) weights to bf16, transposed to [N,K] (one-time; L2 resident after)
    castT(qkv_w,  qkv_wT,  D, 3 * D);
    castT(proj_w, proj_wT, D, D);
    castT(mlp_w1, mlp_w1T, D, 4 * D);
    castT(mlp_w2, mlp_w2T, 4 * D, D);

    // 2) LN1 -> xn (bf16), pad rows zero
    ln_bf16_kernel<<<Mpad, 256, 0, stream>>>(x, ln1_g, ln1_b, xn, M, D);

    // 3) QKV GEMM: [Mpad,768] @ [768,2304] + bias -> f32
    gemm_wmma_kernel<<<dim3(3 * D / GEMM_BN, Mpad / GEMM_BM), 256, 0, stream>>>(
        xn, qkv_wT, qkv_b, nullptr, qkvbuf, nullptr, Mpad, D, 3 * D, 0);

    // 4) attention -> attnb (bf16)
    attn_cls_kernel<<<Bn * H, 128, 0, stream>>>(qkvbuf, attnb, S, H);
    {
        long waves = (long)Bn * H * P;               // 49152
        long blocks = (waves * 32 + 255) / 256;      // 6144
        attn_routed_kernel<<<(int)blocks, 256, 0, stream>>>(qkvbuf, routes, attnb, Bn, S, H, P);
    }
    {   // zero attn pad rows [M, Mpad)
        long n = (long)(Mpad - M) * D;
        zero_bf16_kernel<<<(int)((n + 255) / 256), 256, 0, stream>>>(attnb + (long)M * D, n);
    }

    // 5) proj GEMM: attn @ proj_w + bias + x -> x1 (f32); guard rows < M
    gemm_wmma_kernel<<<dim3(D / GEMM_BN, Mpad / GEMM_BM), 256, 0, stream>>>(
        attnb, proj_wT, proj_b, x, x1, nullptr, M, D, D, 0);

    // 6) LN2 -> h2in (bf16), pad rows zero
    ln_bf16_kernel<<<Mpad, 256, 0, stream>>>(x1, ln2_g, ln2_b, h2in, M, D);

    // 7) MLP1 GEMM: [Mpad,768] @ [768,3072] + bias, GELU -> h1 (bf16)
    gemm_wmma_kernel<<<dim3(4 * D / GEMM_BN, Mpad / GEMM_BM), 256, 0, stream>>>(
        h2in, mlp_w1T, mlp_b1, nullptr, nullptr, h1, Mpad, D, 4 * D, 1);

    // 8) MLP2 GEMM: [Mpad,3072] @ [3072,768] + bias + x1 -> d_out (f32, M rows)
    gemm_wmma_kernel<<<dim3(D / GEMM_BN, Mpad / GEMM_BM), 256, 0, stream>>>(
        h1, mlp_w2T, mlp_b2, x1, out, nullptr, M, 4 * D, D, 0);
}
